// BaseModel_33732673143112
// MI455X (gfx1250) — matrix operand order repaired and checked by reading, not compiled, not run on hardware
//
#include <hip/hip_runtime.h>
#include <cmath>

#define LLEN 512

typedef _Float16 h16;
typedef __attribute__((ext_vector_type(16))) _Float16 v16h;
typedef __attribute__((ext_vector_type(8)))  float    v8f;

struct Off3 { long long mod; long long divs; int div; };
__device__ __host__ inline long long offn(int n, const Off3 o) {
  return (long long)(n % o.div) * o.mod + (long long)(n / o.div) * o.divs;
}

// ---------------------------------------------------------------------------
// Generic WMMA GEMM: C[m,n] (+bias[m]) = scale * sum_k A[m,k]*B[k,n]
// Block = 4 waves sharing one 16-row M tile, covering a 64-wide N strip.
// Each K chunk of 32 is cooperatively staged into LDS in WMMA *fragment
// order*, so each lane's v16h operand is one aligned 32B LDS read
// (ds_load_b128 x2).  Gathers are branchless (clamp + select).  Conv im2col
// uses power-of-two padded kernel windows: ci = kk>>kshift, ko = kk&kmask.
// ---------------------------------------------------------------------------
struct GemmArgs {
  const h16* A; Off3 aOff; int a_rs, a_ks, mBound;
  const h16* B; Off3 bOff; int b_ks, b_ns;
  int is_conv, kshift, kmask, conv_pad;
  int kTotal;
  const float* bias;
  float scale;
  float* C; h16* C16; Off3 cOff; int c_rs;
};

__global__ __launch_bounds__(128) void k_gemm(GemmArgs g) {
  __shared__ __align__(32) h16 As[512];    // 16x32 A chunk, fragment order
  __shared__ __align__(32) h16 Bs[2048];   // 4 x (32x16) B chunks, fragment order
  const int tid  = threadIdx.x;
  const int lane = tid & 31;
  const int wv   = tid >> 5;
  const int nb = blockIdx.x;
  const int mt = blockIdx.y << 4;
  const int n0 = blockIdx.z << 6;
  const long long aoff = offn(nb, g.aOff);
  const long long boff = offn(nb, g.bOff);
  const long long coff = offn(nb, g.cOff);
  v8f acc = {0.f, 0.f, 0.f, 0.f, 0.f, 0.f, 0.f, 0.f};

  for (int k0 = 0; k0 < g.kTotal; k0 += 32) {
    if (!g.is_conv && k0 + 32 < g.kTotal)
      __builtin_prefetch(&g.B[boff + (long long)(k0 + 32) * g.b_ks], 0, 1);
    // ---- stage A: 512 halves, 4 per thread, fragment order ----------------
#pragma unroll
    for (int r = 0; r < 4; ++r) {
      const int e   = tid * 4 + r;
      const int ln  = e >> 4;
      const int idx = e & 15;
      // A layout (16-bit 16x32): kk = (idx&7) + ((idx>>3)<<4) + ((ln>=16)*8)
      const int m  = mt + (ln & 15);
      const int kk = k0 + (idx & 7) + ((idx >> 3) << 4) + ((ln >> 4) << 3);
      const bool ok = (kk < g.kTotal) & (m < g.mBound);
      const long long off = aoff + (long long)m * g.a_rs + (long long)kk * g.a_ks;
      h16 v = g.A[ok ? off : aoff];
      As[e] = ok ? v : (h16)0.f;
    }
    // ---- stage B: 2048 halves, 16 per thread, fragment order --------------
#pragma unroll
    for (int r = 0; r < 16; ++r) {
      const int e   = tid * 16 + r;
      const int w2  = e >> 9;
      const int ln  = (e >> 4) & 31;
      const int idx = e & 15;
      // B layout (16-bit 32x16): kk = (idx&7) + ((idx>>3)<<4) + ((ln>=16)*8), n = ln&15
      const int n  = n0 + (w2 << 4) + (ln & 15);
      const int kk = k0 + (idx & 7) + ((idx >> 3) << 4) + ((ln >> 4) << 3);
      bool ok = (kk < g.kTotal);
      long long off;
      if (g.is_conv) {
        const int ci = kk >> g.kshift;
        const int ko = kk & g.kmask;
        const int l  = n + ko - g.conv_pad;
        ok = ok & (l >= 0) & (l < LLEN);
        off = boff + ((long long)ci << 9) + l;
      } else {
        off = boff + (long long)kk * g.b_ks + (long long)n * g.b_ns;
      }
      h16 v = g.B[ok ? off : boff];
      Bs[e] = ok ? v : (h16)0.f;
    }
    __syncthreads();
    const v16h av = *(const v16h*)&As[lane << 4];
    const v16h bv = *(const v16h*)&Bs[(wv << 9) + (lane << 4)];
    acc = __builtin_amdgcn_wmma_f32_16x16x32_f16(false, av, false, bv,
                                                 (short)0, acc, false, false);
    __syncthreads();
  }

  const int nl = n0 + (wv << 4) + (lane & 15);
#pragma unroll
  for (int p = 0; p < 8; ++p) {
    const int m = mt + p + ((lane >> 4) << 3);  // C/D: m = p + (lane>=16)*8
    if (m < g.mBound) {
      float v = acc[p] * g.scale;
      if (g.bias) v += g.bias[m];
      const long long o = coff + (long long)m * g.c_rs + nl;
      g.C[o] = v;
      if (g.C16) g.C16[o] = (h16)v;
    }
  }
}

// ---------------------------------------------------------------------------
// Channel LayerNorm (+optional pre-add skip, leaky-relu, post-add skip).
// ---------------------------------------------------------------------------
struct LnArgs {
  const float* In; Off3 iOff;
  const float* Pre; Off3 pOff;
  const float* g; const float* b;
  int C; int leaky;
  const float* Post; Off3 qOff;
  float* Out; h16* Out16; Off3 oOff;
};

__global__ __launch_bounds__(256) void k_ln(LnArgs a) {
  const int n = blockIdx.x, l = blockIdx.y, tid = threadIdx.x;
  __shared__ float rs[256], rq[256];
  const long long io = offn(n, a.iOff);
  const long long po = a.Pre ? offn(n, a.pOff) : 0;
  const long long qo = a.Post ? offn(n, a.qOff) : 0;
  const long long oo = offn(n, a.oOff);
  float s = 0.f, s2 = 0.f;
  for (int c = tid; c < a.C; c += 256) {
    float v = a.In[io + (long long)c * LLEN + l];
    if (a.Pre) v += a.Pre[po + (long long)c * LLEN + l];
    s += v; s2 += v * v;
  }
  rs[tid] = s; rq[tid] = s2; __syncthreads();
  for (int o2 = 128; o2 > 0; o2 >>= 1) {
    if (tid < o2) { rs[tid] += rs[tid + o2]; rq[tid] += rq[tid + o2]; }
    __syncthreads();
  }
  const float invC = 1.f / (float)a.C;
  const float mu = rs[0] * invC;
  const float var = rq[0] * invC - mu * mu;
  const float rinv = rsqrtf(var + 1e-5f);
  for (int c = tid; c < a.C; c += 256) {
    float v = a.In[io + (long long)c * LLEN + l];
    if (a.Pre) v += a.Pre[po + (long long)c * LLEN + l];
    float y = (v - mu) * rinv * a.g[c] + a.b[c];
    if (a.leaky) y = y > 0.f ? y : 0.01f * y;
    if (a.Post) y += a.Post[qo + (long long)c * LLEN + l];
    long long o = oo + (long long)c * LLEN + l;
    a.Out[o] = y;
    if (a.Out16) a.Out16[o] = (h16)y;
  }
}

__global__ __launch_bounds__(256) void k_softmax(const float* S, h16* P) {
  const long long row = blockIdx.x;
  const int tid = threadIdx.x;
  const float* s = S + row * LLEN;
  h16* p = P + row * LLEN;
  __shared__ float red[256];
  float m = -1e30f;
  for (int j = tid; j < LLEN; j += 256) m = fmaxf(m, s[j]);
  red[tid] = m; __syncthreads();
  for (int o = 128; o > 0; o >>= 1) { if (tid < o) red[tid] = fmaxf(red[tid], red[tid + o]); __syncthreads(); }
  m = red[0]; __syncthreads();
  float sum = 0.f;
  for (int j = tid; j < LLEN; j += 256) sum += __expf(s[j] - m);
  red[tid] = sum; __syncthreads();
  for (int o = 128; o > 0; o >>= 1) { if (tid < o) red[tid] += red[tid + o]; __syncthreads(); }
  const float inv = 1.f / red[0];
  for (int j = tid; j < LLEN; j += 256) p[j] = (h16)(__expf(s[j] - m) * inv);
}

__global__ void k_w2h(const float* s, h16* d, long long n) {
  long long t = (long long)blockIdx.x * 256 + threadIdx.x;
  if (t < n) d[t] = (h16)s[t];
}
__global__ void k_copyf(const float* s, float* d, long long n) {
  long long t = (long long)blockIdx.x * 256 + threadIdx.x;
  if (t < n) d[t] = s[t];
}
// repack conv weights [O][I][K] -> f16 [O][I][Kp] with zero padding (Kp = 2^s)
__global__ void k_wpack(const float* s, h16* d, int O, int I, int K, int Kp) {
  long long t = (long long)blockIdx.x * 256 + threadIdx.x;
  const long long tot = (long long)O * I * Kp;
  if (t >= tot) return;
  int k = (int)(t % Kp); long long r = t / Kp;
  int ci = (int)(r % I); int o = (int)(r / I);
  d[t] = (k < K) ? (h16)s[((long long)o * I + ci) * K + k] : (h16)0.f;
}
__global__ void k_node(const float* node, float* xf, h16* xh) {
  long long t = (long long)blockIdx.x * 256 + threadIdx.x;
  const long long tot = 16LL * 39 * LLEN;
  if (t >= tot) return;
  int l = (int)(t % LLEN); long long r = t / LLEN;
  int c = (int)(r % 39); int b = (int)(r / 39);
  float v = node[((long long)b * LLEN + l) * 39 + c];
  xf[t] = v; xh[t] = (h16)v;
}
// build [6, B, j, i] f16 adjacency (raw ch 0..4 + learned relu ch 5), stored
// transposed so the einsum 'bij,bcj->bci' becomes a plain row-major GEMM.
__global__ void k_adj(const float* adj, const float* w, const float* b0, h16* adjT) {
  long long t = (long long)blockIdx.x * 256 + threadIdx.x;
  const long long tot = 16LL * LLEN * LLEN;
  if (t >= tot) return;
  int j = (int)(t % LLEN); long long r = t / LLEN;
  int i = (int)(r % LLEN); int bb = (int)(r / LLEN);
  const float* ap = adj + t * 5;
  float acc = b0[0];
#pragma unroll
  for (int c = 0; c < 5; ++c) {
    float v = ap[c];
    adjT[(((long long)c * 16 + bb) * LLEN + j) * LLEN + i] = (h16)v;
    acc += v * w[c];
  }
  adjT[(((long long)5 * 16 + bb) * LLEN + j) * LLEN + i] = (h16)fmaxf(acc, 0.f);
}
__global__ void k_copy_out(const float* src, float* dst, int C, int chOff) {
  long long t = (long long)blockIdx.x * 256 + threadIdx.x;
  const long long tot = 16LL * C * LLEN;
  if (t >= tot) return;
  int l = (int)(t % LLEN); long long r = t / LLEN;
  int c = (int)(r % C); int b = (int)(r / C);
  dst[((long long)b * 135 + chOff + c) * LLEN + l] = src[t];
}

// ---------------------------------------------------------------------------
extern "C" void kernel_launch(void* const* d_in, const int* in_sizes, int n_in,
                              void* d_out, int out_size, void* d_ws, size_t ws_size,
                              hipStream_t stream) {
  (void)in_sizes; (void)n_in; (void)out_size; (void)ws_size;
  // setup_inputs() insertion order: node, adj, then params (recursive insertion
  // order): adj_w, adj_b, f1..f4, f5_64, f5_32, f6_64, f6_32, aa64, aa32,
  // mha64, mha32.  _fp order: cw,cb,ng,nb,rcw,rcb,rng,rnb.  _hp: qw,qb,kw,kb,vw,vb.
  const float* node = (const float*)d_in[0];
  const float* adj  = (const float*)d_in[1];
  auto P = [&](int i) { return (const float*)d_in[i]; };

  char* base = (char*)d_ws;
  size_t cur = 0;
  auto allocF = [&](size_t n) { float* p = (float*)(base + cur); cur += ((n * 4 + 255) / 256) * 256; return p; };
  auto allocH = [&](size_t n) { h16* p = (h16*)(base + cur); cur += ((n * 2 + 255) / 256) * 256; return p; };

  auto W2H = [&](const float* s, h16* d, long long n) {
    k_w2h<<<dim3((unsigned)((n + 255) / 256)), dim3(256), 0, stream>>>(s, d, n);
  };
  auto CPF = [&](const float* s, float* d, long long n) {
    k_copyf<<<dim3((unsigned)((n + 255) / 256)), dim3(256), 0, stream>>>(s, d, n);
  };

  struct FPW { const float *cw,*cb,*ng,*nb,*rcw,*rcb,*rng,*rnb; h16 *wh,*rwh;
               int I,O,K,kp,ksh; };
  auto mkfp = [&](int idx, int I, int O, int K) {
    FPW f; f.cw=P(idx); f.cb=P(idx+1); f.ng=P(idx+2); f.nb=P(idx+3);
    f.rcw=P(idx+4); f.rcb=P(idx+5); f.rng=P(idx+6); f.rnb=P(idx+7);
    f.I=I; f.O=O; f.K=K;
    int kp = 1, ksh = 0;
    while (kp < K) { kp <<= 1; ++ksh; }
    f.kp = kp; f.ksh = ksh;
    f.wh = allocH((size_t)O*I*kp); f.rwh = allocH((size_t)O*O*kp);
    long long t1 = (long long)O*I*kp, t2 = (long long)O*O*kp;
    k_wpack<<<dim3((unsigned)((t1+255)/256)), dim3(256), 0, stream>>>(f.cw,  f.wh,  O, I, K, kp);
    k_wpack<<<dim3((unsigned)((t2+255)/256)), dim3(256), 0, stream>>>(f.rcw, f.rwh, O, O, K, kp);
    return f;
  };
  FPW f1   = mkfp(4, 39, 128, 3);
  FPW f2   = mkfp(12, 128, 64, 7);
  FPW f3   = mkfp(20, 64, 32, 17);
  FPW f4   = mkfp(28, 32, 16, 31);
  FPW f564 = mkfp(36, 240, 64, 31);
  FPW f532 = mkfp(44, 64, 32, 31);
  FPW f664 = mkfp(52, 448, 64, 3);
  FPW f632 = mkfp(60, 224, 32, 3);
  FPW a64f1 = mkfp(68, 240, 64, 3), a64f2 = mkfp(76, 64, 64, 3), a64f3 = mkfp(84, 128, 64, 3);
  FPW a32f1 = mkfp(92, 64, 32, 3),  a32f2 = mkfp(100, 32, 32, 3), a32f3 = mkfp(108, 64, 32, 3);

  struct MHAW { h16 *Wq,*Wk,*Wv; float *Bq,*Bk,*Bv; const float *ng,*nb; int u,d; };
  auto mkmha = [&](int hb, int u) {
    MHAW m; m.u = u; m.d = u / 4;
    m.Wq = allocH((size_t)u*u); m.Wk = allocH((size_t)u*u); m.Wv = allocH((size_t)u*u);
    m.Bq = allocF(u); m.Bk = allocF(u); m.Bv = allocF(u);
    m.ng = P(hb + 24); m.nb = P(hb + 25);
    for (int h = 0; h < 4; ++h) {
      int hbase = hb + 6 * h;   // qw,qb,kw,kb,vw,vb
      W2H(P(hbase+0), m.Wq + (size_t)h*m.d*u, (long long)m.d*u);
      W2H(P(hbase+2), m.Wk + (size_t)h*m.d*u, (long long)m.d*u);
      W2H(P(hbase+4), m.Wv + (size_t)h*m.d*u, (long long)m.d*u);
      CPF(P(hbase+1), m.Bq + h*m.d, m.d);
      CPF(P(hbase+3), m.Bk + h*m.d, m.d);
      CPF(P(hbase+5), m.Bv + h*m.d, m.d);
    }
    return m;
  };
  MHAW mha64 = mkmha(116, 64);
  MHAW mha32 = mkmha(142, 32);

  // activations / scratch
  h16*   adjT = allocH((size_t)6*16*LLEN*LLEN);
  float* x0f  = allocF((size_t)16*39*LLEN);  h16* x0h  = allocH((size_t)16*39*LLEN);
  float* XCf  = allocF((size_t)16*240*LLEN); h16* XCh  = allocH((size_t)16*240*LLEN);
  float* T1   = allocF((size_t)96*64*LLEN);
  float* Hf   = allocF((size_t)96*64*LLEN);  h16* Hh   = allocH((size_t)96*64*LLEN);
  float* FB1f = allocF((size_t)16*64*LLEN);  h16* FB1h = allocH((size_t)16*64*LLEN);
  float* X12f = allocF((size_t)96*128*LLEN); h16* X12h = allocH((size_t)96*128*LLEN);
  float* FB2f = allocF((size_t)96*64*LLEN);  h16* FB2h = allocH((size_t)96*64*LLEN);
  float* CATf = allocF((size_t)16*448*LLEN); h16* CATh = allocH((size_t)16*448*LLEN);
  float* F6f  = allocF((size_t)16*64*LLEN);  h16* F6h  = allocH((size_t)16*64*LLEN);
  float* Qf   = allocF((size_t)16*64*LLEN);  h16* Qh   = allocH((size_t)16*64*LLEN);
  float* Kf   = allocF((size_t)16*64*LLEN);  h16* Kh   = allocH((size_t)16*64*LLEN);
  float* Vf   = allocF((size_t)16*64*LLEN);  h16* Vh   = allocH((size_t)16*64*LLEN);
  float* Sf   = allocF((size_t)16*4*LLEN*LLEN);
  h16*   Ph   = allocH((size_t)16*4*LLEN*LLEN);
  float* ATTf = allocF((size_t)16*64*LLEN);
  float* M64f = allocF((size_t)16*64*LLEN);  h16* M64h = allocH((size_t)16*64*LLEN);
  float* M32f = allocF((size_t)16*32*LLEN);

  auto GEMM = [&](const GemmArgs& g, int nBatch, int Mtiles) {
    k_gemm<<<dim3(nBatch, Mtiles, 8), dim3(128), 0, stream>>>(g);
  };
  auto LN = [&](const LnArgs& a, int nBatch) {
    k_ln<<<dim3(nBatch, LLEN), dim3(256), 0, stream>>>(a);
  };

  // full fwd_block: conv -> LN+leaky, conv -> LN+leaky, + skip
  auto FWD = [&](const FPW& f, const h16* in16, Off3 inO,
                 float* outF, h16* outH, Off3 outO, int nBatch) {
    Off3 lin1{(long long)f.O * LLEN, 0, nBatch};
    GemmArgs g1{};
    g1.A = f.wh; g1.aOff = Off3{0, 0, 1}; g1.a_rs = f.I * f.kp; g1.a_ks = 1; g1.mBound = f.O;
    g1.B = in16; g1.bOff = inO; g1.b_ks = 0; g1.b_ns = 0;
    g1.is_conv = 1; g1.kshift = f.ksh; g1.kmask = f.kp - 1; g1.conv_pad = f.K / 2;
    g1.kTotal = f.I * f.kp;
    g1.bias = f.cb; g1.scale = 1.f;
    g1.C = T1; g1.C16 = nullptr; g1.cOff = lin1; g1.c_rs = LLEN;
    GEMM(g1, nBatch, f.O / 16);
    LnArgs l1{};
    l1.In = T1; l1.iOff = lin1; l1.Pre = nullptr; l1.g = f.ng; l1.b = f.nb;
    l1.C = f.O; l1.leaky = 1; l1.Post = nullptr;
    l1.Out = Hf; l1.Out16 = Hh; l1.oOff = lin1;
    LN(l1, nBatch);
    GemmArgs g2 = g1;
    g2.A = f.rwh; g2.a_rs = f.O * f.kp; g2.kTotal = f.O * f.kp;
    g2.B = Hh; g2.bOff = lin1; g2.bias = f.rcb;
    GEMM(g2, nBatch, f.O / 16);
    LnArgs l2{};
    l2.In = T1; l2.iOff = lin1; l2.Pre = nullptr; l2.g = f.rng; l2.b = f.rnb;
    l2.C = f.O; l2.leaky = 1; l2.Post = Hf; l2.qOff = lin1;
    l2.Out = outF; l2.Out16 = outH; l2.oOff = outO;
    LN(l2, nBatch);
  };

  // ---- stage 0 -------------------------------------------------------------
  { long long tot = 16LL*39*LLEN;
    k_node<<<dim3((unsigned)((tot+255)/256)), dim3(256), 0, stream>>>(node, x0f, x0h); }
  { long long tot = 16LL*LLEN*LLEN;
    k_adj<<<dim3((unsigned)((tot+255)/256)), dim3(256), 0, stream>>>(adj, P(2), P(3), adjT); }

  Off3 xcO{240LL * LLEN, 0, 16};
  FWD(f1, x0h, Off3{39LL * LLEN, 0, 16}, XCf, XCh, xcO, 16);
  FWD(f2, XCh,                     xcO, XCf + (size_t)128*LLEN, XCh + (size_t)128*LLEN, xcO, 16);
  FWD(f3, XCh + (size_t)128*LLEN,  xcO, XCf + (size_t)192*LLEN, XCh + (size_t)192*LLEN, xcO, 16);
  FWD(f4, XCh + (size_t)192*LLEN,  xcO, XCf + (size_t)224*LLEN, XCh + (size_t)224*LLEN, xcO, 16);

  // ---- stage loop: unit = 64 then 32 --------------------------------------
  const h16* xin16 = XCh; Off3 xinO = xcO;
  float* Mf[2] = {M64f, M32f}; h16* Mh[2] = {M64h, nullptr};
  for (int si = 0; si < 2; ++si) {
    const int u = si == 0 ? 64 : 32;
    const int d = u / 4;
    const FPW &af1 = si ? a32f1 : a64f1, &af2 = si ? a32f2 : a64f2, &af3 = si ? a32f3 : a64f3;
    const FPW &f5 = si ? f532 : f564, &f6 = si ? f632 : f664;
    const MHAW &mp = si ? mha32 : mha64;

    Off3 fb1O{(long long)u * LLEN, 0, 16};
    FWD(af1, xin16, xinO, FB1f, FB1h, fb1O, 16);

    Off3 x12O{2LL * u * LLEN, 0, 96};
    Off3 adjO{(long long)LLEN * LLEN, 0, 96};
    GemmArgs gx{};   // X1 = a @ FB1^T  (n = ch*16 + b)
    gx.A = FB1h; gx.aOff = Off3{(long long)u * LLEN, 0, 16}; gx.a_rs = LLEN; gx.a_ks = 1; gx.mBound = u;
    gx.B = adjT; gx.bOff = adjO; gx.b_ks = LLEN; gx.b_ns = 1;
    gx.is_conv = 0; gx.kshift = 0; gx.kmask = 0; gx.conv_pad = 0; gx.kTotal = LLEN;
    gx.bias = nullptr; gx.scale = 1.f;
    gx.C = X12f; gx.C16 = X12h; gx.cOff = x12O; gx.c_rs = LLEN;
    GEMM(gx, 96, u / 16);

    FWD(af2, X12h, x12O, FB2f, FB2h, Off3{(long long)u * LLEN, 0, 96}, 96);

    GemmArgs gx2 = gx;   // X2 = a @ FB2^T
    gx2.A = FB2h; gx2.aOff = Off3{(long long)u * LLEN, 0, 96};
    gx2.C = X12f + (size_t)u * LLEN; gx2.C16 = X12h + (size_t)u * LLEN;
    GEMM(gx2, 96, u / 16);

    Off3 catO{7LL * u * LLEN, (long long)u * LLEN, 16};   // -> CAT[b][ch*u + c][l]
    FWD(af3, X12h, x12O, CATf, CATh, catO, 96);
    Off3 cat5O{7LL * u * LLEN, 0, 16};
    FWD(f5, xin16, xinO, CATf + (size_t)6*u*LLEN, CATh + (size_t)6*u*LLEN, cat5O, 16);

    Off3 f6O{(long long)u * LLEN, 0, 16};
    FWD(f6, CATh, cat5O, F6f, F6h, f6O, 16);

    auto qkv = [&](h16* W, float* Bia, float* Of, h16* Oh) {  // K=1 conv as GEMM
      GemmArgs q{};
      q.A = W; q.aOff = Off3{0, 0, 1}; q.a_rs = u; q.a_ks = 1; q.mBound = u;
      q.B = F6h; q.bOff = f6O; q.b_ks = LLEN; q.b_ns = 1;
      q.is_conv = 0; q.kshift = 0; q.kmask = 0; q.conv_pad = 0; q.kTotal = u;
      q.bias = Bia; q.scale = 1.f;
      q.C = Of; q.C16 = Oh; q.cOff = f6O; q.c_rs = LLEN;
      GEMM(q, 16, u / 16);
    };
    qkv(mp.Wq, mp.Bq, Qf, Qh);
    qkv(mp.Wk, mp.Bk, Kf, Kh);
    qkv(mp.Wv, mp.Bv, Vf, Vh);

    Off3 headO{(long long)d * LLEN, (long long)u * LLEN, 4};  // n = b*4 + h
    GemmArgs gs{};   // S[i,j] = sum_dd Q[dd,i]K[dd,j] / sqrt(d)
    gs.A = Qh; gs.aOff = headO; gs.a_rs = 1; gs.a_ks = LLEN; gs.mBound = LLEN;
    gs.B = Kh; gs.bOff = headO; gs.b_ks = LLEN; gs.b_ns = 1;
    gs.is_conv = 0; gs.kshift = 0; gs.kmask = 0; gs.conv_pad = 0; gs.kTotal = d;
    gs.bias = nullptr; gs.scale = 1.f / sqrtf((float)d);
    gs.C = Sf; gs.C16 = nullptr; gs.cOff = Off3{(long long)LLEN * LLEN, 0, 64}; gs.c_rs = LLEN;
    GEMM(gs, 64, 32);

    k_softmax<<<dim3(64 * LLEN), dim3(256), 0, stream>>>(Sf, Ph);

    GemmArgs ga{};   // ATT[dd,i] = sum_j V[dd,j] P[i,j]
    ga.A = Vh; ga.aOff = headO; ga.a_rs = LLEN; ga.a_ks = 1; ga.mBound = d;
    ga.B = Ph; ga.bOff = Off3{(long long)LLEN * LLEN, 0, 64}; ga.b_ks = 1; ga.b_ns = LLEN;
    ga.is_conv = 0; ga.kshift = 0; ga.kmask = 0; ga.conv_pad = 0; ga.kTotal = LLEN;
    ga.bias = nullptr; ga.scale = 1.f;
    ga.C = ATTf; ga.C16 = nullptr; ga.cOff = headO; ga.c_rs = LLEN;
    GEMM(ga, 64, 1);

    LnArgs lm{};   // LN(x + att) over channels, gamma/beta, no leaky
    lm.In = F6f; lm.iOff = f6O; lm.Pre = ATTf; lm.pOff = f6O;
    lm.g = mp.ng; lm.b = mp.nb; lm.C = u; lm.leaky = 0; lm.Post = nullptr;
    lm.Out = Mf[si]; lm.Out16 = Mh[si]; lm.oOff = f6O;
    LN(lm, 16);

    { long long tot = 16LL * u * LLEN;
      k_copy_out<<<dim3((unsigned)((tot+255)/256)), dim3(256), 0, stream>>>(
          Mf[si], (float*)d_out, u, si == 0 ? 39 : 103); }

    xin16 = M64h; xinO = Off3{64LL * LLEN, 0, 16};
  }

  { long long tot = 16LL * 39 * LLEN;
    k_copy_out<<<dim3((unsigned)((tot+255)/256)), dim3(256), 0, stream>>>(
        x0f, (float*)d_out, 39, 0); }
}
